// LSTMBrockett_32195074851281
// MI455X (gfx1250) — compile-verified, matrix operand order checked by reading, not run on hardware
//
#include <hip/hip_runtime.h>
#include <hip/hip_bf16.h>
#include <math.h>

#define B_SZ 512
#define T_SZ 256
#define I_SZ 64
#define H_SZ 512
#define NG   2048      // 4*H
#define K_SZ 576       // I + H (fused [x_t | h] K dimension)
#define HN   30
#define EPS  1e-5f

typedef __attribute__((ext_vector_type(16))) __bf16 v16bf;
typedef __attribute__((ext_vector_type(8)))  float  v8f;

// ---------- helpers ----------

__device__ __forceinline__ unsigned short f2bf(float f) {
  // round-to-nearest-even fp32 -> bf16 (bit pattern)
  unsigned u = __float_as_uint(f);
  return (unsigned short)((u + 0x7FFFu + ((u >> 16) & 1u)) >> 16);
}

union FragU { uint4 q[2]; v16bf v; };

// Load a 16x32 bf16 WMMA A/B fragment from a row-major [rows][row_stride]
// bf16 buffer. Per ISA 7.12.2 (16-bit A 16x32): lane holds row (lane&15);
// lanes 0-15 carry K = {0..7, 16..23}, lanes 16-31 carry K = {8..15, 24..31}.
// => per lane: two contiguous 8-element (16B) chunks at k0 + 8*(lane>>4) and +16.
__device__ __forceinline__ v16bf load_frag(const unsigned short* __restrict__ base,
                                           int row_stride, int row0, int k0, int lane) {
  int r = row0 + (lane & 15);
  int k = k0 + ((lane >> 4) << 3);
  const unsigned short* p = base + (size_t)r * row_stride + k;
  FragU u;
  u.q[0] = *(const uint4*)p;
  u.q[1] = *(const uint4*)(p + 16);
  return u.v;
}

#define WMMA_BF16(A, Bf, C) \
  __builtin_amdgcn_wmma_f32_16x16x32_bf16(false, (A), false, (Bf), (short)0, (C), false, false)

__device__ __forceinline__ float sigf(float x) { return 1.f / (1.f + expf(-x)); }

// ---------- one-time prep kernels ----------

__global__ void k_cvt_x(const float* __restrict__ x, unsigned short* __restrict__ xbf, int n) {
  int i = blockIdx.x * blockDim.x + threadIdx.x;
  if (i < n) xbf[i] = f2bf(x[i]);
}

// wcomb[n][k] : k<64 -> W_ih[n][k], else W_hh[n][k-64]   (bf16, [2048][576])
__global__ void k_prep_w(const float* __restrict__ wih, const float* __restrict__ whh,
                         unsigned short* __restrict__ wcomb) {
  int i = blockIdx.x * blockDim.x + threadIdx.x;
  if (i >= NG * K_SZ) return;
  int n = i / K_SZ, k = i - n * K_SZ;
  float v = (k < I_SZ) ? wih[(size_t)n * I_SZ + k] : whh[(size_t)n * H_SZ + (k - I_SZ)];
  wcomb[i] = f2bf(v);
}

__global__ void k_prep_bias(const float* __restrict__ bih, const float* __restrict__ bhh,
                            float* __restrict__ bias) {
  int i = blockIdx.x * blockDim.x + threadIdx.x;
  if (i < NG) bias[i] = bih[i] + bhh[i];
}

// l1wbf: [32][512] bf16, rows >= 30 zero-padded
__global__ void k_prep_l1w(const float* __restrict__ l1w, unsigned short* __restrict__ l1wbf) {
  int i = blockIdx.x * blockDim.x + threadIdx.x;
  if (i >= 32 * H_SZ) return;
  int r = i >> 9;
  l1wbf[i] = (r < HN) ? f2bf(l1w[(size_t)r * H_SZ + (i & 511)]) : (unsigned short)0;
}

__global__ void k_zero_state(float* __restrict__ c, unsigned short* __restrict__ hbf) {
  int i = blockIdx.x * blockDim.x + threadIdx.x;
  if (i < B_SZ * H_SZ) { c[i] = 0.f; hbf[i] = 0; }
}

// ---------- per-step: fused-K recurrent GEMM  gates = [x_t|h] @ Wcomb^T + bias ----------
// grid (8, 16), 256 threads = 8 waves; wave tile 32x32; WG tile 64x128.
__global__ __launch_bounds__(256) void k_gates(
    const unsigned short* __restrict__ xbf,   // [B][T][I] bf16
    const unsigned short* __restrict__ hbf,   // [B][H]    bf16
    const unsigned short* __restrict__ wcomb, // [2048][576] bf16
    const float* __restrict__ bias,           // [2048]
    float* __restrict__ gates,                // [B][2048]
    int t) {
  const int lane = threadIdx.x & 31;
  const int w    = threadIdx.x >> 5;                  // 0..7
  const int m0   = blockIdx.x * 64 + (w & 1) * 32;    // batch rows
  const int n0   = blockIdx.y * 128 + (w >> 1) * 32;  // gate cols

  v8f acc[2][2] = {};

  const unsigned short* xrow = xbf + (size_t)t * I_SZ;  // + b*(T*I) via stride

  for (int kc = 0; kc < K_SZ; kc += 32) {
    // prefetch next weight chunk (emits global_prefetch_b8)
    if (kc + 32 < K_SZ)
      __builtin_prefetch(wcomb + (size_t)(n0 + (lane & 15)) * K_SZ + kc + 32, 0, 3);

    v16bf a0, a1;
    if (kc < I_SZ) {  // x region of fused K
      a0 = load_frag(xrow, T_SZ * I_SZ, m0,      kc, lane);
      a1 = load_frag(xrow, T_SZ * I_SZ, m0 + 16, kc, lane);
    } else {          // h region
      a0 = load_frag(hbf, H_SZ, m0,      kc - I_SZ, lane);
      a1 = load_frag(hbf, H_SZ, m0 + 16, kc - I_SZ, lane);
    }
    v16bf b0 = load_frag(wcomb, K_SZ, n0,      kc, lane);
    v16bf b1 = load_frag(wcomb, K_SZ, n0 + 16, kc, lane);

    acc[0][0] = WMMA_BF16(a0, b0, acc[0][0]);
    acc[0][1] = WMMA_BF16(a0, b1, acc[0][1]);
    acc[1][0] = WMMA_BF16(a1, b0, acc[1][0]);
    acc[1][1] = WMMA_BF16(a1, b1, acc[1][1]);
  }

  #pragma unroll
  for (int mi = 0; mi < 2; ++mi)
    #pragma unroll
    for (int ni = 0; ni < 2; ++ni) {
      int col   = n0 + ni * 16 + (lane & 15);
      float bv  = bias[col];
      int rbase = m0 + mi * 16 + ((lane >> 4) << 3);
      #pragma unroll
      for (int rr = 0; rr < 8; ++rr)
        gates[(size_t)(rbase + rr) * NG + col] = acc[mi][ni][rr] + bv;
    }
}

// ---------- per-step: LSTM cell elementwise (full grid) ----------
__global__ __launch_bounds__(256) void k_cell(const float* __restrict__ gates,
                                              float* __restrict__ c,
                                              unsigned short* __restrict__ hbf) {
  int idx = blockIdx.x * blockDim.x + threadIdx.x;  // 0 .. B*H-1
  if (idx >= B_SZ * H_SZ) return;
  int b = idx >> 9, j = idx & 511;
  const float* g = gates + (size_t)b * NG;
  float ig = sigf(g[j]);
  float fg = sigf(g[H_SZ + j]);
  float gg = tanhf(g[2 * H_SZ + j]);
  float og = sigf(g[3 * H_SZ + j]);
  float cn = fg * c[idx] + ig * gg;
  float hn = og * tanhf(cn);
  c[idx]   = cn;
  hbf[idx] = f2bf(hn);
}

// ---------- per-step: fused MLP head, ONE workgroup (512 threads = 16 waves) ----------
// Phase 1: WMMA GEMM  a1 = h @ l1_w^T + l1_b  -> LDS [512][32]
// Phase 2: 4x (batch BN + ReLU) + 30x30 chain + final [2,30] projection.
__device__ __forceinline__ void bn_relu(float* a, float* s_sum, float* s_sq,
                                        const float* __restrict__ gamma,
                                        const float* __restrict__ beta) {
  __syncthreads();
  if (threadIdx.x < HN) { s_sum[threadIdx.x] = 0.f; s_sq[threadIdx.x] = 0.f; }
  __syncthreads();
  #pragma unroll
  for (int j = 0; j < HN; ++j) {
    atomicAdd(&s_sum[j], a[j]);
    atomicAdd(&s_sq[j], a[j] * a[j]);
  }
  __syncthreads();
  #pragma unroll
  for (int j = 0; j < HN; ++j) {
    float m = s_sum[j] * (1.f / (float)B_SZ);
    float v = s_sq[j] * (1.f / (float)B_SZ) - m * m;       // biased var
    float x = gamma[j] * (a[j] - m) * rsqrtf(v + EPS) + beta[j];
    a[j] = x > 0.f ? x : 0.f;
  }
}

__global__ __launch_bounds__(512) void k_mlp(const unsigned short* __restrict__ hbf,   // [512][512] bf16
                                             const unsigned short* __restrict__ l1wbf, // [32][512] bf16
                                             const float* __restrict__ l1b,
                                             const float* __restrict__ l2w,
                                             const float* __restrict__ l3w,
                                             const float* __restrict__ l4w,
                                             const float* __restrict__ l5w,  // [2][30]
                                             const float* __restrict__ gamma,
                                             const float* __restrict__ beta,
                                             float* __restrict__ out, int t) {
  __shared__ float s_a1[B_SZ * 32];   // 64 KB (of 320 KB LDS)
  __shared__ float s_sum[32];
  __shared__ float s_sq[32];

  // ---- phase 1: l1 GEMM via WMMA; wave w owns rows [w*32, w*32+32) ----
  {
    const int lane = threadIdx.x & 31;
    const int w    = threadIdx.x >> 5;     // 0..15
    const int m0   = w * 32;

    v8f acc[2][2] = {};
    for (int kc = 0; kc < H_SZ; kc += 32) {
      v16bf a0  = load_frag(hbf, H_SZ, m0,      kc, lane);
      v16bf a1f = load_frag(hbf, H_SZ, m0 + 16, kc, lane);
      v16bf b0  = load_frag(l1wbf, H_SZ, 0,  kc, lane);
      v16bf b1  = load_frag(l1wbf, H_SZ, 16, kc, lane);
      acc[0][0] = WMMA_BF16(a0,  b0, acc[0][0]);
      acc[0][1] = WMMA_BF16(a0,  b1, acc[0][1]);
      acc[1][0] = WMMA_BF16(a1f, b0, acc[1][0]);
      acc[1][1] = WMMA_BF16(a1f, b1, acc[1][1]);
    }

    #pragma unroll
    for (int mi = 0; mi < 2; ++mi)
      #pragma unroll
      for (int ni = 0; ni < 2; ++ni) {
        int col   = ni * 16 + (lane & 15);
        float bv  = (col < HN) ? l1b[col] : 0.f;
        int rbase = m0 + mi * 16 + ((lane >> 4) << 3);
        #pragma unroll
        for (int rr = 0; rr < 8; ++rr)
          s_a1[(rbase + rr) * 32 + col] = acc[mi][ni][rr] + bv;
      }
  }
  __syncthreads();

  // ---- phase 2: per-thread batch row through BN/ReLU chain ----
  const int b = threadIdx.x;
  float a[HN], r[HN];
  #pragma unroll
  for (int j = 0; j < HN; ++j) a[j] = s_a1[b * 32 + j];

  bn_relu(a, s_sum, s_sq, gamma, beta);

  const float* lw[3] = { l2w, l3w, l4w };
  for (int L = 0; L < 3; ++L) {
    const float* wmat = lw[L];
    for (int j = 0; j < HN; ++j) {
      float s = 0.f;
      #pragma unroll
      for (int k = 0; k < HN; ++k) s = fmaf(a[k], wmat[j * HN + k], s);
      r[j] = s;
    }
    bn_relu(r, s_sum, s_sq, gamma, beta);
    #pragma unroll
    for (int j = 0; j < HN; ++j) a[j] = r[j];
  }

  float o0 = 0.f, o1 = 0.f;
  #pragma unroll
  for (int k = 0; k < HN; ++k) {
    o0 = fmaf(a[k], l5w[k], o0);
    o1 = fmaf(a[k], l5w[HN + k], o1);
  }
  // out shape [B, 2, T]
  out[(size_t)b * (2 * T_SZ) + t]        = o0;
  out[(size_t)b * (2 * T_SZ) + T_SZ + t] = o1;
}

// ---------- host-side launch ----------

extern "C" void kernel_launch(void* const* d_in, const int* in_sizes, int n_in,
                              void* d_out, int out_size, void* d_ws, size_t ws_size,
                              hipStream_t stream) {
  const float* x    = (const float*)d_in[0];
  const float* Wih  = (const float*)d_in[1];
  const float* Whh  = (const float*)d_in[2];
  const float* bih  = (const float*)d_in[3];
  const float* bhh  = (const float*)d_in[4];
  const float* l1w  = (const float*)d_in[5];
  const float* l1b  = (const float*)d_in[6];
  const float* l2w  = (const float*)d_in[7];
  const float* l3w  = (const float*)d_in[8];
  const float* l4w  = (const float*)d_in[9];
  const float* l5w  = (const float*)d_in[10];
  const float* gam  = (const float*)d_in[11];
  const float* bet  = (const float*)d_in[12];
  float* out = (float*)d_out;
  (void)in_sizes; (void)n_in; (void)out_size; (void)ws_size;

  char* ws = (char*)d_ws;
  size_t off = 0;
  auto take = [&](size_t bytes) { char* p = ws + off; off += (bytes + 255) & ~(size_t)255; return p; };

  unsigned short* xbf   = (unsigned short*)take((size_t)B_SZ * T_SZ * I_SZ * 2); // 16.8 MB
  unsigned short* wcomb = (unsigned short*)take((size_t)NG * K_SZ * 2);          // 2.36 MB
  float*          bias  = (float*)take((size_t)NG * 4);
  float*          gates = (float*)take((size_t)B_SZ * NG * 4);                   // 4 MB
  float*          cbuf  = (float*)take((size_t)B_SZ * H_SZ * 4);
  unsigned short* hbf   = (unsigned short*)take((size_t)B_SZ * H_SZ * 2);
  unsigned short* l1wbf = (unsigned short*)take((size_t)32 * H_SZ * 2);

  const int nx = B_SZ * T_SZ * I_SZ;
  k_cvt_x    <<<(nx + 255) / 256, 256, 0, stream>>>(x, xbf, nx);
  k_prep_w   <<<(NG * K_SZ + 255) / 256, 256, 0, stream>>>(Wih, Whh, wcomb);
  k_prep_bias<<<(NG + 255) / 256, 256, 0, stream>>>(bih, bhh, bias);
  k_prep_l1w <<<(32 * H_SZ + 255) / 256, 256, 0, stream>>>(l1w, l1wbf);
  k_zero_state<<<(B_SZ * H_SZ + 255) / 256, 256, 0, stream>>>(cbuf, hbf);

  for (int t = 0; t < T_SZ; ++t) {
    k_gates<<<dim3(B_SZ / 64, NG / 128), 256, 0, stream>>>(xbf, hbf, wcomb, bias, gates, t);
    k_cell <<<(B_SZ * H_SZ + 255) / 256, 256, 0, stream>>>(gates, cbuf, hbf);
    k_mlp  <<<1, 512, 0, stream>>>(hbf, l1wbf, l1b, l2w, l3w, l4w, l5w, gam, bet, out, t);
  }
}
